// TransformerBlock_5317169513215
// MI455X (gfx1250) — matrix operand order repaired and checked by reading, not compiled, not run on hardware
//
#include <hip/hip_runtime.h>
#include <hip/hip_bf16.h>

#define D_MODEL  1024
#define N_HEADS  16
#define HEAD_DIM 64
#define SEQ      2048
#define BATCH    4
#define HIDDEN   4096
#define ROWS     (BATCH * SEQ)   // 8192

typedef __attribute__((ext_vector_type(8)))  unsigned short ushort8;
typedef __attribute__((ext_vector_type(16))) unsigned short ushort16;
typedef __attribute__((ext_vector_type(16))) __bf16        bf16x16;
typedef __attribute__((ext_vector_type(8)))  float         f32x8;

// ---------------------------------------------------------------- helpers
__device__ __forceinline__ unsigned short f32_to_bf16(float f) {
  unsigned int u = __float_as_uint(f);
  u += 0x7FFFu + ((u >> 16) & 1u);        // round-to-nearest-even
  return (unsigned short)(u >> 16);
}
__device__ __forceinline__ float bf16_to_f32(unsigned short h) {
  return __uint_as_float(((unsigned int)h) << 16);
}

// Async DMA of 16B global -> LDS (CDNA5 GLOBAL_LOAD_ASYNC_TO_LDS_B128,
// tracked by ASYNCcnt).  LDS offset = low 32 bits of the flat shared pointer
// (generic->LDS aperture truncation per ISA 10.2).
__device__ __forceinline__ void async_ld_b128(const unsigned short* gptr,
                                              unsigned short* lptr) {
  unsigned loff = (unsigned)(uintptr_t)lptr;
  unsigned long long ga = (unsigned long long)(uintptr_t)gptr;
  asm volatile("global_load_async_to_lds_b128 %0, %1, off"
               :: "v"(loff), "v"(ga) : "memory");
}
__device__ __forceinline__ void wait_asynccnt0() {
  asm volatile("s_wait_asynccnt 0" ::: "memory");
}

// 16-bit A-matrix fragment (16x32, MxK).  Per ISA: lane m = lane%16,
// element e -> k = 16*(e/8) + 8*(lane/16) + (e%8).  Two contiguous 16B runs.
__device__ __forceinline__ bf16x16 frag_ld_A(const unsigned short* rowbase, int hf) {
  ushort8 lo = *(const ushort8*)(rowbase + 8 * hf);
  ushort8 hi = *(const ushort8*)(rowbase + 16 + 8 * hf);
  union { ushort16 v; ushort8 h[2]; } t;
  t.h[0] = lo; t.h[1] = hi;
  return __builtin_bit_cast(bf16x16, t.v);
}
// 16-bit B-matrix fragment (32x16, KxN) read from an LDS tile stored as
// [n][k]: lane n = lane%16, element e -> k = 16*(e/8) + 8*(lane/16) + (e%8).
__device__ __forceinline__ bf16x16 frag_ld_B(const unsigned short* rowbase, int hf) {
  ushort8 lo = *(const ushort8*)(rowbase + 8 * hf);
  ushort8 hi = *(const ushort8*)(rowbase + 16 + 8 * hf);
  union { ushort16 v; ushort8 h[2]; } t;
  t.h[0] = lo; t.h[1] = hi;
  return __builtin_bit_cast(bf16x16, t.v);
}
__device__ __forceinline__ f32x8 v8zero() {
  f32x8 z;
#pragma unroll
  for (int i = 0; i < 8; ++i) z[i] = 0.0f;
  return z;
}
__device__ __forceinline__ f32x8 wmma_bf16(bf16x16 a, bf16x16 b, f32x8 c) {
  return __builtin_amdgcn_wmma_f32_16x16x32_bf16(false, a, false, b,
                                                 (short)0, c, false, false);
}

// ---------------------------------------------------------------- elementwise
__global__ void cvt_f32_bf16_kernel(const float* __restrict__ in,
                                    unsigned short* __restrict__ out, int n) {
  int i = blockIdx.x * blockDim.x + threadIdx.x;
  if (i < n) out[i] = f32_to_bf16(in[i]);
}

__global__ void gatemul_kernel(const float* __restrict__ gate,
                               const float* __restrict__ o,
                               unsigned short* __restrict__ out, int n) {
  int i = blockIdx.x * blockDim.x + threadIdx.x;
  if (i < n) out[i] = f32_to_bf16(gate[i] * o[i]);
}

// one block per row of 1024
__global__ __launch_bounds__(256)
void ln_kernel(const float* __restrict__ x, const float* __restrict__ w,
               const float* __restrict__ b, unsigned short* __restrict__ out) {
  __shared__ float rs[256], rs2[256];
  const int row = blockIdx.x, tid = threadIdx.x;
  const float* xr = x + (size_t)row * D_MODEL;
  float s = 0.f, s2 = 0.f;
  for (int i = tid; i < D_MODEL; i += 256) { float v = xr[i]; s += v; s2 += v * v; }
  rs[tid] = s; rs2[tid] = s2; __syncthreads();
  for (int off = 128; off > 0; off >>= 1) {
    if (tid < off) { rs[tid] += rs[tid + off]; rs2[tid] += rs2[tid + off]; }
    __syncthreads();
  }
  const float mu  = rs[0] * (1.0f / D_MODEL);
  const float var = rs2[0] * (1.0f / D_MODEL) - mu * mu;
  const float inv = rsqrtf(var + 1e-5f);
  for (int i = tid; i < D_MODEL; i += 256)
    out[(size_t)row * D_MODEL + i] = f32_to_bf16((xr[i] - mu) * inv * w[i] + b[i]);
}

// in-place RoPE on bf16 (B,S,H,Dh) laid out as rows=B*S, cols=D
__global__ void rope_kernel(unsigned short* __restrict__ q) {
  int idx = blockIdx.x * blockDim.x + threadIdx.x;    // ROWS * H * (Dh/2)
  int row = idx >> 9;                                 // / 512
  int rem = idx & 511;
  int h   = rem >> 5;
  int i   = rem & 31;
  int s   = row & (SEQ - 1);
  float freq = __powf(10000.0f, -(float)(2 * i) / (float)HEAD_DIM);
  float sn, cs;
  __sincosf((float)s * freq, &sn, &cs);
  unsigned short* p = q + (size_t)row * D_MODEL + h * HEAD_DIM + 2 * i;
  float xr = bf16_to_f32(p[0]);
  float xi = bf16_to_f32(p[1]);
  p[0] = f32_to_bf16(xr * cs - xi * sn);
  p[1] = f32_to_bf16(xr * sn + xi * cs);
}

// ---------------------------------------------------------------- GEMM (bf16 WMMA)
// C[M,N] = A[M,K] * B[K,N] + bias, epilogue per EPI.
// Block tile 128x128x32, 256 threads = 8 waves (2M x 4N), wave tile 64x32
// (4x2 WMMA, 8 v_wmma per K-step from 6 fragment loads).
enum { EPI_BF16 = 0, EPI_SIGMOID_F32 = 1, EPI_SILU_BF16 = 2, EPI_RESID_F32 = 3 };

#define GEMM_TM 128
#define GEMM_TN 128
#define GEMM_TK 32
#define LDA_P 40   // padded K-stride (halfwords): 80B rows, 16B aligned
#define LDB_P 40

template <int EPI>
__global__ __launch_bounds__(256)
void gemm_bf16_kernel(const unsigned short* __restrict__ A,
                      const unsigned short* __restrict__ B,
                      const float* __restrict__ bias,
                      const float* __restrict__ resid,
                      float* __restrict__ outf,
                      unsigned short* __restrict__ outb,
                      int M, int N, int K) {
  __shared__ unsigned short sA[GEMM_TM * LDA_P];   // [m][k]
  __shared__ unsigned short sB[GEMM_TN * LDB_P];   // [n][k]  (transposed)

  const int tid  = threadIdx.x;
  const int lane = tid & 31, wid = tid >> 5;
  const int hf   = lane >> 4, l16 = lane & 15;
  const int wm   = wid >> 2,  wn  = wid & 3;       // 2M x 4N wave grid
  const int m0   = blockIdx.y * GEMM_TM;
  const int n0   = blockIdx.x * GEMM_TN;

  f32x8 acc[4][2];
#pragma unroll
  for (int i = 0; i < 4; ++i)
#pragma unroll
    for (int j = 0; j < 2; ++j) acc[i][j] = v8zero();

  const int arow = tid >> 1;          // 0..127
  const int acol = (tid & 1) * 16;    // 0 or 16
  const int brow = tid >> 3;          // k 0..31
  const int bcol = (tid & 7) * 16;    // n 0..112

  for (int k0 = 0; k0 < K; k0 += GEMM_TK) {
    // stage A tile via async DMA directly into LDS (no VGPR bounce)
    {
      const unsigned short* g = A + (size_t)(m0 + arow) * K + k0 + acol;
      async_ld_b128(g,     &sA[arow * LDA_P + acol]);
      async_ld_b128(g + 8, &sA[arow * LDA_P + acol + 8]);
      if (k0 + GEMM_TK < K)
        __builtin_prefetch(g + GEMM_TK, 0, 0);   // global_prefetch_b8
    }
    // stage B tile transposed -> sB[n][k] (needs transpose: register path)
    {
      const unsigned short* g = B + (size_t)(k0 + brow) * N + n0 + bcol;
      ushort8 v0 = *(const ushort8*)g;
      ushort8 v1 = *(const ushort8*)(g + 8);
#pragma unroll
      for (int j = 0; j < 8; ++j) {
        sB[(bcol + j) * LDB_P + brow]     = v0[j];
        sB[(bcol + 8 + j) * LDB_P + brow] = v1[j];
      }
    }
    wait_asynccnt0();   // A-tile DMA landed in LDS
    __syncthreads();

    bf16x16 b0 = frag_ld_B(&sB[(32 * wn +      l16) * LDB_P], hf);
    bf16x16 b1 = frag_ld_B(&sB[(32 * wn + 16 + l16) * LDB_P], hf);
#pragma unroll
    for (int mi = 0; mi < 4; ++mi) {
      bf16x16 a = frag_ld_A(&sA[(64 * wm + 16 * mi + l16) * LDA_P], hf);
      acc[mi][0] = wmma_bf16(a, b0, acc[mi][0]);
      acc[mi][1] = wmma_bf16(a, b1, acc[mi][1]);
    }
    __syncthreads();
  }

  // epilogue: C-layout is row M = r + 8*(lane/16), col N = lane%16
#pragma unroll
  for (int mi = 0; mi < 4; ++mi) {
#pragma unroll
    for (int ni = 0; ni < 2; ++ni) {
      const int rowb = m0 + 64 * wm + 16 * mi + 8 * hf;
      const int col  = n0 + 32 * wn + 16 * ni + l16;
      const float bs = bias[col];
#pragma unroll
      for (int r = 0; r < 8; ++r) {
        const size_t idx = (size_t)(rowb + r) * N + col;
        float v = acc[mi][ni][r] + bs;
        if (EPI == EPI_BF16) {
          outb[idx] = f32_to_bf16(v);
        } else if (EPI == EPI_SIGMOID_F32) {
          outf[idx] = 1.0f / (1.0f + __expf(-v));
        } else if (EPI == EPI_SILU_BF16) {
          outb[idx] = f32_to_bf16(v / (1.0f + __expf(-v)));
        } else {  // EPI_RESID_F32
          outf[idx] = resid[idx] + v;
        }
      }
    }
  }
}

// ---------------------------------------------------------------- flash attention
// grid (S/64, H, B), 128 threads = 4 waves; 64-query block, 64-key tiles.
#define ATT_P 72   // padded stride (halfwords): 144B rows, 16B aligned

__global__ __launch_bounds__(128)
void attn_kernel(const unsigned short* __restrict__ qb,
                 const unsigned short* __restrict__ kb,
                 const unsigned short* __restrict__ vb,
                 float* __restrict__ O) {
  __shared__ unsigned short lq[64 * ATT_P];   // [query][dh]
  __shared__ unsigned short lk[64 * ATT_P];   // [key][dh]
  __shared__ unsigned short lv[64 * ATT_P];   // [dh][key] (transposed)
  __shared__ unsigned short lp[64 * ATT_P];   // [query][key]

  const int tid  = threadIdx.x;
  const int lane = tid & 31, wid = tid >> 5;     // 4 waves, 16 queries each
  const int hf   = lane >> 4, l16 = lane & 15;
  const int qt = blockIdx.x, h = blockIdx.y, b = blockIdx.z;
  const size_t rbase = (size_t)(b * SEQ);
  const int    cbase = h * HEAD_DIM;

  // stage Q block (64x64) via async DMA
  {
    const int row = tid >> 1, cb = (tid & 1) * 32;
    const unsigned short* g = qb + (rbase + qt * 64 + row) * D_MODEL + cbase + cb;
#pragma unroll
    for (int j = 0; j < 4; ++j)
      async_ld_b128(g + 8 * j, &lq[row * ATT_P + cb + 8 * j]);
    wait_asynccnt0();
  }
  __syncthreads();

  bf16x16 qf[2];
#pragma unroll
  for (int ks = 0; ks < 2; ++ks)
    qf[ks] = frag_ld_A(&lq[(16 * wid + l16) * ATT_P + 32 * ks], hf);

  f32x8 o[4];
#pragma unroll
  for (int n = 0; n < 4; ++n) o[n] = v8zero();
  float m_r[8], l_r[8];
#pragma unroll
  for (int r = 0; r < 8; ++r) { m_r[r] = -3.0e38f; l_r[r] = 0.0f; }

  for (int kt = 0; kt <= qt; ++kt) {
    __syncthreads();   // all waves done with previous K/V tiles
    // stage K tile [key][dh] (async DMA) + V tile transposed [dh][key]
    {
      const int key = tid >> 1, cb = (tid & 1) * 32;
      const unsigned short* gk = kb + (rbase + kt * 64 + key) * D_MODEL + cbase + cb;
      const unsigned short* gv = vb + (rbase + kt * 64 + key) * D_MODEL + cbase + cb;
#pragma unroll
      for (int j = 0; j < 4; ++j) {
        async_ld_b128(gk + 8 * j, &lk[key * ATT_P + cb + 8 * j]);
        ushort8 vv = *(const ushort8*)(gv + 8 * j);
#pragma unroll
        for (int i = 0; i < 8; ++i) lv[(cb + 8 * j + i) * ATT_P + key] = vv[i];
      }
      wait_asynccnt0();
    }
    __syncthreads();

    // S = Q * K^T   (16 queries x 64 keys per wave)
    f32x8 s[4];
#pragma unroll
    for (int j = 0; j < 4; ++j) {
      s[j] = v8zero();
#pragma unroll
      for (int ks = 0; ks < 2; ++ks) {
        bf16x16 kf = frag_ld_B(&lk[(16 * j + l16) * ATT_P + 32 * ks], hf);
        s[j] = wmma_bf16(qf[ks], kf, s[j]);
      }
    }

    // scale + causal mask + online softmax
    float sv[4][8];
#pragma unroll
    for (int j = 0; j < 4; ++j) {
      const int kj = kt * 64 + 16 * j + l16;
#pragma unroll
      for (int r = 0; r < 8; ++r) {
        const int qi = qt * 64 + 16 * wid + r + 8 * hf;
        float v = s[j][r] * 0.125f;           // 1/sqrt(64)
        sv[j][r] = (kj > qi) ? -1.0e30f : v;
      }
    }
#pragma unroll
    for (int r = 0; r < 8; ++r) {
      float t = sv[0][r];
#pragma unroll
      for (int j = 1; j < 4; ++j) t = fmaxf(t, sv[j][r]);
      t = fmaxf(t, __shfl_xor(t, 8, 16));
      t = fmaxf(t, __shfl_xor(t, 4, 16));
      t = fmaxf(t, __shfl_xor(t, 2, 16));
      t = fmaxf(t, __shfl_xor(t, 1, 16));
      const float mnew  = fmaxf(m_r[r], t);
      const float alpha = __expf(m_r[r] - mnew);
      float lsum = 0.0f;
#pragma unroll
      for (int j = 0; j < 4; ++j) {
        float p = __expf(sv[j][r] - mnew);
        sv[j][r] = p;
        lsum += p;
      }
      lsum += __shfl_xor(lsum, 8, 16);
      lsum += __shfl_xor(lsum, 4, 16);
      lsum += __shfl_xor(lsum, 2, 16);
      lsum += __shfl_xor(lsum, 1, 16);
      l_r[r] = l_r[r] * alpha + lsum;
      m_r[r] = mnew;
#pragma unroll
      for (int n = 0; n < 4; ++n) o[n][r] *= alpha;
      // P -> LDS (wave-private rows) for C-layout -> A-layout conversion
#pragma unroll
      for (int j = 0; j < 4; ++j)
        lp[(16 * wid + r + 8 * hf) * ATT_P + 16 * j + l16] = f32_to_bf16(sv[j][r]);
    }

    // O += P * V
#pragma unroll
    for (int ks = 0; ks < 2; ++ks) {
      bf16x16 pa = frag_ld_A(&lp[(16 * wid + l16) * ATT_P + 32 * ks], hf);
#pragma unroll
      for (int n = 0; n < 4; ++n) {
        bf16x16 vf = frag_ld_B(&lv[(16 * n + l16) * ATT_P + 32 * ks], hf);
        o[n] = wmma_bf16(pa, vf, o[n]);
      }
    }
  }

  // normalize + write O (fp32, (B,S,H,Dh) == rows x D layout)
#pragma unroll
  for (int r = 0; r < 8; ++r) {
    const float inv = 1.0f / l_r[r];
    const size_t row = rbase + qt * 64 + 16 * wid + r + 8 * hf;
#pragma unroll
    for (int n = 0; n < 4; ++n)
      O[row * D_MODEL + cbase + 16 * n + l16] = o[n][r] * inv;
  }
}

// ---------------------------------------------------------------- launch
extern "C" void kernel_launch(void* const* d_in, const int* in_sizes, int n_in,
                              void* d_out, int out_size, void* d_ws, size_t ws_size,
                              hipStream_t stream) {
  (void)in_sizes; (void)n_in; (void)out_size; (void)ws_size;
  const float* x    = (const float*)d_in[0];
  const float* Wq   = (const float*)d_in[1];
  const float* Wk   = (const float*)d_in[2];
  const float* Wv   = (const float*)d_in[3];
  const float* Wg   = (const float*)d_in[4];
  const float* Wo   = (const float*)d_in[5];
  const float* bq   = (const float*)d_in[6];
  const float* bk   = (const float*)d_in[7];
  const float* bv   = (const float*)d_in[8];
  const float* bg   = (const float*)d_in[9];
  const float* bo   = (const float*)d_in[10];
  const float* ln1w = (const float*)d_in[11];
  const float* ln1b = (const float*)d_in[12];
  const float* ln2w = (const float*)d_in[13];
  const float* ln2b = (const float*)d_in[14];
  const float* W1   = (const float*)d_in[15];
  const float* b1   = (const float*)d_in[16];
  const float* W2   = (const float*)d_in[17];
  const float* b2   = (const float*)d_in[18];

  char* ws = (char*)d_ws;
  size_t off = 0;
  auto alloc = [&](size_t bytes) -> void* {
    void* p = ws + off;
    off += (bytes + 255) & ~(size_t)255;
    return p;
  };
  const size_t RD = (size_t)ROWS * D_MODEL;
  unsigned short* wq_b  = (unsigned short*)alloc((size_t)1024 * 1024 * 2);
  unsigned short* wk_b  = (unsigned short*)alloc((size_t)1024 * 1024 * 2);
  unsigned short* wv_b  = (unsigned short*)alloc((size_t)1024 * 1024 * 2);
  unsigned short* wg_b  = (unsigned short*)alloc((size_t)1024 * 1024 * 2);
  unsigned short* wo_b  = (unsigned short*)alloc((size_t)1024 * 1024 * 2);
  unsigned short* w1_b  = (unsigned short*)alloc((size_t)1024 * 4096 * 2);
  unsigned short* w2_b  = (unsigned short*)alloc((size_t)4096 * 1024 * 2);
  unsigned short* xn_b  = (unsigned short*)alloc(RD * 2);
  unsigned short* q_b   = (unsigned short*)alloc(RD * 2);
  unsigned short* k_b   = (unsigned short*)alloc(RD * 2);
  unsigned short* v_b   = (unsigned short*)alloc(RD * 2);
  float*          gatef = (float*)alloc(RD * 4);
  float*          Of    = (float*)alloc(RD * 4);
  unsigned short* go_b  = (unsigned short*)alloc(RD * 2);
  float*          x1f   = (float*)alloc(RD * 4);
  unsigned short* xn2_b = (unsigned short*)alloc(RD * 2);
  unsigned short* h_b   = (unsigned short*)alloc((size_t)ROWS * HIDDEN * 2);

  // 1. weights -> bf16
  {
    int n = 1024 * 1024, g = (n + 255) / 256;
    cvt_f32_bf16_kernel<<<g, 256, 0, stream>>>(Wq, wq_b, n);
    cvt_f32_bf16_kernel<<<g, 256, 0, stream>>>(Wk, wk_b, n);
    cvt_f32_bf16_kernel<<<g, 256, 0, stream>>>(Wv, wv_b, n);
    cvt_f32_bf16_kernel<<<g, 256, 0, stream>>>(Wg, wg_b, n);
    cvt_f32_bf16_kernel<<<g, 256, 0, stream>>>(Wo, wo_b, n);
    int n2 = 1024 * 4096, g2 = (n2 + 255) / 256;
    cvt_f32_bf16_kernel<<<g2, 256, 0, stream>>>(W1, w1_b, n2);
    cvt_f32_bf16_kernel<<<g2, 256, 0, stream>>>(W2, w2_b, n2);
  }
  // 2. xn = LN1(x)
  ln_kernel<<<ROWS, 256, 0, stream>>>(x, ln1w, ln1b, xn_b);
  // 3. Q/K/V/gate projections
  dim3 gqkv(D_MODEL / GEMM_TN, ROWS / GEMM_TM);
  gemm_bf16_kernel<EPI_BF16><<<gqkv, 256, 0, stream>>>(xn_b, wq_b, bq, nullptr, nullptr, q_b, ROWS, D_MODEL, D_MODEL);
  gemm_bf16_kernel<EPI_BF16><<<gqkv, 256, 0, stream>>>(xn_b, wk_b, bk, nullptr, nullptr, k_b, ROWS, D_MODEL, D_MODEL);
  gemm_bf16_kernel<EPI_BF16><<<gqkv, 256, 0, stream>>>(xn_b, wv_b, bv, nullptr, nullptr, v_b, ROWS, D_MODEL, D_MODEL);
  gemm_bf16_kernel<EPI_SIGMOID_F32><<<gqkv, 256, 0, stream>>>(xn_b, wg_b, bg, nullptr, gatef, nullptr, ROWS, D_MODEL, D_MODEL);
  // 4. RoPE on Q and K
  {
    int np = ROWS * N_HEADS * (HEAD_DIM / 2);
    rope_kernel<<<np / 256, 256, 0, stream>>>(q_b);
    rope_kernel<<<np / 256, 256, 0, stream>>>(k_b);
  }
  // 5. causal flash attention
  attn_kernel<<<dim3(SEQ / 64, N_HEADS, BATCH), 128, 0, stream>>>(q_b, k_b, v_b, Of);
  // 6. gate * attn_out -> bf16
  gatemul_kernel<<<(int)(RD / 256), 256, 0, stream>>>(gatef, Of, go_b, (int)RD);
  // 7. x1 = x + (gate*out) @ Wo + bo
  gemm_bf16_kernel<EPI_RESID_F32><<<gqkv, 256, 0, stream>>>(go_b, wo_b, bo, x, x1f, nullptr, ROWS, D_MODEL, D_MODEL);
  // 8. xn2 = LN2(x1)
  ln_kernel<<<ROWS, 256, 0, stream>>>(x1f, ln2w, ln2b, xn2_b);
  // 9. h = silu(xn2 @ W1 + b1)
  dim3 gff(HIDDEN / GEMM_TN, ROWS / GEMM_TM);
  gemm_bf16_kernel<EPI_SILU_BF16><<<gff, 256, 0, stream>>>(xn2_b, w1_b, b1, nullptr, nullptr, h_b, ROWS, HIDDEN, D_MODEL);
  // 10. out = x1 + h @ W2 + b2
  gemm_bf16_kernel<EPI_RESID_F32><<<gqkv, 256, 0, stream>>>(h_b, w2_b, b2, x1f, (float*)d_out, nullptr, ROWS, D_MODEL, HIDDEN);
}
